// SWMSA_44418551775955
// MI455X (gfx1250) — matrix operand order repaired and checked by reading, compiled, not run on hardware
//
#include <hip/hip_runtime.h>

typedef __attribute__((ext_vector_type(16))) __bf16 v16bf;
typedef __attribute__((ext_vector_type(8)))  __bf16 v8bf;
typedef __attribute__((ext_vector_type(8)))  float  v8f;

#define WMMA_BF16(a, b, c) \
  __builtin_amdgcn_wmma_f32_16x16x32_bf16(false, (a), false, (b), (short)0, (c), false, false)

static __device__ __forceinline__ v16bf cat8(v8bf lo, v8bf hi) {
  return __builtin_shufflevector(lo, hi, 0,1,2,3,4,5,6,7,8,9,10,11,12,13,14,15);
}

static __device__ __forceinline__ int region8(int c) {
  // Swin shift-mask region id along one axis: H=64, ws=8, shift=4
  return (c < 56) ? 0 : ((c < 60) ? 1 : 2);
}

// ---------------------------------------------------------------------------
// Kernel 1: repack weight matrix (K x N, f32 row-major) into fragment-major
// bf16 B-tiles: dst[(nt*8+kt)*32 + lane][16]; lane holds col n = lane&15,
// K = kt*32 + (lane>>4)*16 + 0..15 (contiguous along K).
// ---------------------------------------------------------------------------
__global__ void swin_pack_w(const float* __restrict__ w, __bf16* __restrict__ dst, int ldn) {
  const int t    = blockIdx.x * blockDim.x + threadIdx.x;
  const int lane = t & 31;
  const int kt   = (t >> 5) & 7;
  const int nt   = t >> 8;
  const int n    = nt * 16 + (lane & 15);
  const int k0   = kt * 32 + (lane >> 4) * 16;
  __bf16* d = dst + ((size_t)((nt * 8 + kt) * 32 + lane) << 4);
#pragma unroll
  for (int j = 0; j < 16; ++j)
    d[j] = (__bf16)w[(size_t)(k0 + j) * ldn + n];
}

// ---------------------------------------------------------------------------
// Kernel 2: fused shift + window-partition + QKV GEMM (M=131072, K=256, N=768).
// One block (4 waves) owns a 16-token M-tile and all 48 N-tiles (12 per wave).
// The f32 A-tile is gathered from x ONCE per block, converted to bf16, staged
// in LDS in WMMA fragment layout (double-buffered), and shared by all waves.
// Scatter q/k token-major and v transposed (dh-major) for the attention stage.
// ---------------------------------------------------------------------------
__global__ __launch_bounds__(128) void swin_qkv_gemm(
    const float* __restrict__ x, const __bf16* __restrict__ Bw,
    const float* __restrict__ bias,
    __bf16* __restrict__ qb, __bf16* __restrict__ kb, __bf16* __restrict__ vt) {
  __shared__ __align__(16) __bf16 lds_a[2][32][16];   // [buf][lane][frag elem]

  const int tid   = threadIdx.x;
  const int lane  = tid & 31;
  const int wave  = tid >> 5;
  const int lrow  = lane & 15;
  const int lhalf = lane >> 4;
  const int mt    = blockIdx.x;                       // 8192 M-tiles

  // --- loader-side token coordinate (token = mt*16 + (tid&15)) -------------
  const int lm   = mt * 16 + (tid & 15);
  const int lb   = lm >> 12;
  const int lwin = (lm >> 6) & 63;
  const int ln   = lm & 63;
  const int lhs  = (lwin >> 3) * 8 + (ln >> 3);
  const int lws  = (lwin & 7) * 8 + (ln & 7);
  const int lh   = (lhs + 4) & 63;                    // cyclic shift by -4
  const int lw   = (lws + 4) & 63;
  const float* xrow = x + ((size_t)(lb * 256) << 12) + (lh << 6) + lw;
  // loader covers fragment lane = tid&31? No: lane2 token row = tid&15, K-half
  // = (tid>>4)&1; element group j0 = (tid>>5)*4.
  const int lane2 = tid & 31;
  const int j0    = (tid >> 5) * 4;                   // 4 frag elems per thread
  const int khalf = (lane2 >> 4) * 16;
  const int jofs  = 0;                                // frag elems contiguous along K

  v8f acc[12];
  const v8f zero = {0.f, 0.f, 0.f, 0.f, 0.f, 0.f, 0.f, 0.f};
#pragma unroll
  for (int t = 0; t < 12; ++t) acc[t] = zero;

  for (int kt = 0; kt < 8; ++kt) {
    const int buf = kt & 1;
    // stage A fragment tile into LDS (bf16, fragment layout)
    const int chb = kt * 32 + khalf + j0 + jofs;
#pragma unroll
    for (int i = 0; i < 4; ++i)
      lds_a[buf][lane2][j0 + i] = (__bf16)xrow[(size_t)(chb + i) << 12];
    if (kt < 7)
      __builtin_prefetch(&xrow[(size_t)(chb + 32) << 12], 0, 1);
    __syncthreads();

    const v16bf a = *(const v16bf*)lds_a[buf][lane];
#pragma unroll
    for (int t = 0; t < 12; ++t) {
      const int nt = wave * 12 + t;
      const v16bf bf = *(const v16bf*)(Bw + ((size_t)((nt * 8 + kt) * 32 + lane) << 4));
      acc[t] = WMMA_BF16(a, bf, acc[t]);
    }
  }

  // --- epilogue: bias + scatter (q/k token-major, v dh-major) --------------
  // A 16-token tile stays inside one window (16 | 64): hoist decomposition.
  const int tok0 = mt * 16;
  const int tb   = tok0 >> 12;
  const int twin = (tok0 >> 6) & 63;
  const int tnb  = (tok0 & 63) + 8 * lhalf;
  const size_t bw0 = (size_t)(tb * 64 + twin) * 8;    // (b*nWin + win)*heads

#pragma unroll
  for (int t = 0; t < 12; ++t) {
    const int nt   = wave * 12 + t;
    const int outc = nt * 16 + lrow;
    const float bv = bias[outc];
    const int part = nt >> 4;                         // 0=q 1=k 2=v (uniform)
    const int nt2  = nt & 15;
    const int head = nt2 >> 1;                        // uniform per tile
    const int dh   = (nt2 & 1) * 16 + lrow;
    const size_t bwh = bw0 + head;
    if (part == 0) {
#pragma unroll
      for (int r = 0; r < 8; ++r)
        qb[(bwh * 64 + tnb + r) * 32 + dh] = (__bf16)(acc[t][r] + bv);
    } else if (part == 1) {
#pragma unroll
      for (int r = 0; r < 8; ++r)
        kb[(bwh * 64 + tnb + r) * 32 + dh] = (__bf16)(acc[t][r] + bv);
    } else {
#pragma unroll
      for (int r = 0; r < 8; ++r)
        vt[(bwh * 32 + dh) * 64 + tnb + r] = (__bf16)(acc[t][r] + bv);
    }
  }
}

// ---------------------------------------------------------------------------
// Kernel 3: windowed attention. Block = 4 waves = one (b, win, head).
// Wave computes a 16-row strip: S = q k^T (4 WMMA, dh=32 is one K-pass),
// scale + rel-pos bias + analytic shift mask, rowwise softmax (shfl_xor),
// P -> LDS bf16, O = P V (4 WMMA).
// ---------------------------------------------------------------------------
__global__ __launch_bounds__(128) void swin_attn(
    const __bf16* __restrict__ qb, const __bf16* __restrict__ kb,
    const __bf16* __restrict__ vt, const float* __restrict__ relt,
    __bf16* __restrict__ ao) {
  __shared__ __align__(16) __bf16 pbuf[4][16][64];

  const int lane  = threadIdx.x & 31;
  const int wave  = threadIdx.x >> 5;
  const int lrow  = lane & 15;
  const int lhalf = lane >> 4;
  const int head  = blockIdx.x & 7;
  const int win   = (blockIdx.x >> 3) & 63;
  const int b     = blockIdx.x >> 9;

  const size_t base = (size_t)(b * 64 + win) * 8 + head;
  const __bf16* q = qb + base * (64 * 32);
  const __bf16* k = kb + base * (64 * 32);
  const __bf16* v = vt + base * (32 * 64);

  // A fragment: 16 q rows of this wave's strip
  const int am = wave * 16 + lrow;
  const v8bf qlo = *(const v8bf*)(q + am * 32 + lhalf * 16);
  const v8bf qhi = *(const v8bf*)(q + am * 32 + lhalf * 16 + 8);
  const v16bf afrag = cat8(qlo, qhi);

  const v8f zero = {0.f, 0.f, 0.f, 0.f, 0.f, 0.f, 0.f, 0.f};
  v8f s[4];
#pragma unroll
  for (int nt = 0; nt < 4; ++nt) {
    s[nt] = zero;
    const int nn = nt * 16 + lrow;                       // B col = key token
    const v16bf bfrag = *(const v16bf*)(k + nn * 32 + lhalf * 16);
    s[nt] = WMMA_BF16(afrag, bfrag, s[nt]);
  }

  // scale + relative position bias + shift mask
  const int wh = win >> 3, ww = win & 7;
  const float scale = 0.17677669529663687f;              // 1/sqrt(32)
#pragma unroll
  for (int nt = 0; nt < 4; ++nt) {
    const int ncol = nt * 16 + lrow;
    const int ni = ncol >> 3, nj = ncol & 7;
    const int idn = region8(wh * 8 + ni) * 3 + region8(ww * 8 + nj);
#pragma unroll
    for (int r = 0; r < 8; ++r) {
      const int mrow = wave * 16 + r + 8 * lhalf;
      const int mi = mrow >> 3, mj = mrow & 7;
      const float bias = relt[((mi - ni + 7) * 15 + (mj - nj + 7)) * 8 + head];
      const int idm = region8(wh * 8 + mi) * 3 + region8(ww * 8 + mj);
      const float mk = (idm != idn) ? -100.f : 0.f;
      s[nt][r] = s[nt][r] * scale + bias + mk;
    }
  }

  // rowwise softmax: row lives in one 16-lane half across the 4 N-tiles
#pragma unroll
  for (int r = 0; r < 8; ++r) {
    float mx = fmaxf(fmaxf(s[0][r], s[1][r]), fmaxf(s[2][r], s[3][r]));
#pragma unroll
    for (int off = 8; off >= 1; off >>= 1) mx = fmaxf(mx, __shfl_xor(mx, off, 32));
    float sum = 0.f;
#pragma unroll
    for (int nt = 0; nt < 4; ++nt) {
      const float e = __expf(s[nt][r] - mx);
      s[nt][r] = e;
      sum += e;
    }
#pragma unroll
    for (int off = 8; off >= 1; off >>= 1) sum += __shfl_xor(sum, off, 32);
    const float inv = 1.f / sum;
#pragma unroll
    for (int nt = 0; nt < 4; ++nt)
      pbuf[wave][r + 8 * lhalf][nt * 16 + lrow] = (__bf16)(s[nt][r] * inv);
  }

  // O = P @ V   (K = 64 tokens = 2 K-steps, N = dh = 2 tiles)
  v8f o[2];
  o[0] = zero; o[1] = zero;
#pragma unroll
  for (int kt = 0; kt < 2; ++kt) {
    const v8bf plo = *(const v8bf*)&pbuf[wave][lrow][kt * 32 + lhalf * 16];
    const v8bf phi = *(const v8bf*)&pbuf[wave][lrow][kt * 32 + lhalf * 16 + 8];
    const v16bf pa = cat8(plo, phi);
#pragma unroll
    for (int nt = 0; nt < 2; ++nt) {
      const int dh = nt * 16 + lrow;                     // B col = dh index
      const v16bf vb = *(const v16bf*)(v + dh * 64 + kt * 32 + lhalf * 16);
      o[nt] = WMMA_BF16(pa, vb, o[nt]);
    }
  }

  __bf16* aop = ao + (size_t)((b * 64 + win) * 64) * 256 + head * 32;
#pragma unroll
  for (int nt = 0; nt < 2; ++nt)
#pragma unroll
    for (int r = 0; r < 8; ++r) {
      const int tok = wave * 16 + r + 8 * lhalf;
      aop[tok * 256 + nt * 16 + lrow] = (__bf16)o[nt][r];
    }
}

// ---------------------------------------------------------------------------
// Kernel 4: output projection (131072 x 256) @ (256 x 256) + bias, fused
// window-reverse + reverse cyclic shift scatter into (B,C,H,W) f32.
// ---------------------------------------------------------------------------
__global__ __launch_bounds__(128) void swin_oproj(
    const __bf16* __restrict__ A, const __bf16* __restrict__ Bw,
    const float* __restrict__ bias, float* __restrict__ out) {
  const int lane  = threadIdx.x & 31;
  const int wave  = threadIdx.x >> 5;
  const int gw    = blockIdx.x * 4 + wave;
  const int mt    = gw >> 2;           // 8192 M-tiles
  const int ng    = gw & 3;            // 4 groups of 4 N-tiles (16 total)
  const int lrow  = lane & 15;
  const int lhalf = lane >> 4;

  const int m = mt * 16 + lrow;
  const __bf16* arow = A + (size_t)m * 256;

  const v8f zero = {0.f, 0.f, 0.f, 0.f, 0.f, 0.f, 0.f, 0.f};
  v8f acc[4];
#pragma unroll
  for (int t = 0; t < 4; ++t) acc[t] = zero;

  for (int kt = 0; kt < 8; ++kt) {
    const v8bf alo = *(const v8bf*)(arow + kt * 32 + lhalf * 16);
    const v8bf ahi = *(const v8bf*)(arow + kt * 32 + lhalf * 16 + 8);
    const v16bf a = cat8(alo, ahi);
#pragma unroll
    for (int t = 0; t < 4; ++t) {
      const int nt = ng * 4 + t;
      const v16bf bf = *(const v16bf*)(Bw + ((size_t)((nt * 8 + kt) * 32 + lane) << 4));
      acc[t] = WMMA_BF16(a, bf, acc[t]);
    }
  }

  // tile stays inside one window: hoist token decomposition
  const int tok0 = mt * 16;
  const int b    = tok0 >> 12;
  const int win  = (tok0 >> 6) & 63;
  const int nb   = (tok0 & 63) + 8 * lhalf;
  const int whb  = (win >> 3) * 8;
  const int wwb  = (win & 7) * 8;

#pragma unroll
  for (int t = 0; t < 4; ++t) {
    const int nt   = ng * 4 + t;
    const int outc = nt * 16 + lrow;
    const float bv = bias[outc];
    const float* dummy = bias;  (void)dummy;
    float* orow = out + ((size_t)(b * 256 + outc) << 12);
#pragma unroll
    for (int r = 0; r < 8; ++r) {
      const int n  = nb + r;
      const int h  = (whb + (n >> 3) + 4) & 63;   // reverse cyclic shift
      const int w  = (wwb + (n & 7) + 4) & 63;
      orow[(h << 6) + w] = acc[t][r] + bv;
    }
  }
}

// ---------------------------------------------------------------------------
extern "C" void kernel_launch(void* const* d_in, const int* in_sizes, int n_in,
                              void* d_out, int out_size, void* d_ws, size_t ws_size,
                              hipStream_t stream) {
  const float* x     = (const float*)d_in[0];  // (32,256,64,64)
  const float* w_qkv = (const float*)d_in[1];  // (256,768)
  const float* b_qkv = (const float*)d_in[2];  // (768,)
  const float* w_o   = (const float*)d_in[3];  // (256,256)
  const float* b_o   = (const float*)d_in[4];  // (256,)
  const float* relt  = (const float*)d_in[5];  // (225,8)
  float* out = (float*)d_out;

  // workspace carve-up
  char* p = (char*)d_ws;
  __bf16* Bq = (__bf16*)p; p += (size_t)48 * 8 * 32 * 16 * 2;   // 384 KB
  __bf16* Bo = (__bf16*)p; p += (size_t)16 * 8 * 32 * 16 * 2;   // 128 KB
  const size_t TOKC = (size_t)131072 * 256;                     // 33.5M elems
  __bf16* qb = (__bf16*)p; p += TOKC * 2;
  __bf16* kb = (__bf16*)p; p += TOKC * 2;
  __bf16* vt = (__bf16*)p; p += TOKC * 2;
  __bf16* ao = (__bf16*)p; p += TOKC * 2;
  (void)ws_size; (void)out_size; (void)n_in; (void)in_sizes;

  // 1) pack weights into WMMA B-fragment layout (bf16)
  swin_pack_w<<<48, 256, 0, stream>>>(w_qkv, Bq, 768);
  swin_pack_w<<<16, 256, 0, stream>>>(w_o,   Bo, 256);

  // 2) fused shift+partition+QKV GEMM: one block per 16-token M-tile,
  //    block covers all 48 N-tiles; x read exactly once from HBM.
  swin_qkv_gemm<<<8192, 128, 0, stream>>>(x, Bq, b_qkv, qb, kb, vt);

  // 3) attention: one block per (b, win, head) = 32*64*8
  swin_attn<<<16384, 128, 0, stream>>>(qb, kb, vt, relt, ao);

  // 4) output projection + window reverse + unshift: 8192 Mtiles x 4 Ngroups
  swin_oproj<<<8192, 128, 0, stream>>>(ao, Bo, b_o, out);
}
